// DualEdgeHead_4209067950555
// MI455X (gfx1250) — matrix operand order repaired and checked by reading, not compiled, not run on hardware
//
#include <hip/hip_runtime.h>

typedef __attribute__((ext_vector_type(16))) __bf16        v16bf;
typedef __attribute__((ext_vector_type(8)))  float         v8f;
typedef __attribute__((ext_vector_type(4)))  float         v4f;
typedef __attribute__((ext_vector_type(4)))  unsigned int  v4u;

#define NODE_DIM   128
#define CAT        16
#define INPUT_DIM  726
#define KPAD1      736
#define KC1        23
#define CELL_H     192
#define NET_H      128
#define OUT_DIM    128
#define EPS        1e-5f

#define EDGES_PER_BLOCK 32
#define FSTRIDE    744      // bf16 elems per feature row (726 padded, 16B-aligned rows)
#define HCS        193      // float stride, cell hidden
#define HNS        129      // float stride, net hidden
#define A2CS       200      // bf16 stride, relu'd cell hidden (400B rows, 16B aligned)
#define A2NS       136      // bf16 stride, relu'd net hidden  (272B rows, 16B aligned)

// LDS layout (bytes). feat [0,47616) overlays the H/A2 region (disjoint phases).
#define OFF_HC     0
#define OFF_HN     24704    // 32*193*4
#define OFF_A2C    41216    // OFF_HN + 32*129*4
#define OFF_A2N    54016    // OFF_A2C + 32*200*2
#define SMEM_BYTES 62720    // OFF_A2N + 32*136*2

__device__ __forceinline__ int oov(int x, int n) { return (x >= 0 && x < n) ? x : 0; }

__device__ __forceinline__ void st2(__bf16* p, float a, float b) {
    union { __bf16 h[2]; unsigned int u; } x;
    x.h[0] = (__bf16)a; x.h[1] = (__bf16)b;
    *(unsigned int*)p = x.u;
}

// Build a 16-element bf16 fragment from two 16-byte chunks.
__device__ __forceinline__ v16bf load_frag2(const __bf16* p0, const __bf16* p1) {
    union { v16bf v; v4u q[2]; } u;
    u.q[0] = *(const v4u*)p0;
    u.q[1] = *(const v4u*)p1;
    return u.v;
}

__device__ __forceinline__ v8f wmma_bf16(v16bf a, v16bf b, v8f c) {
    return __builtin_amdgcn_wmma_f32_16x16x32_bf16(false, a, false, b, (short)0, c, false, false);
}

// ---------------------------------------------------------------------------
// Pre-pass: convert fp32 W[K][N] (row-major) to bf16 packed in this kernel's WMMA B-operand
// register order (contiguous k, as the A loads use): index ((jt*KC + kc)*32 + lane)*16 + e, where
//   n = jt*16 + (lane&15),  k = kc*32 + (lane>=16 ? 16 : 0) + e,  k>=Kreal -> 0
// ---------------------------------------------------------------------------
__global__ void pack_b_kernel(const float* __restrict__ W, __bf16* __restrict__ outp,
                              int Kreal, int KC, int N) {
    int id = blockIdx.x * 256 + threadIdx.x;
    int NT = N >> 4;
    int total = NT * KC * 32;
    if (id >= total) return;
    int lane = id & 31;
    int rem  = id >> 5;
    int kc   = rem % KC;
    int jt   = rem / KC;
    int ncol = jt * 16 + (lane & 15);
    int kbase = kc * 32 + ((lane >> 4) << 4);
    union { __bf16 h[16]; v4u q[2]; } u;
#pragma unroll
    for (int e2 = 0; e2 < 16; e2++) {
        int k = kbase + e2;
        float v = (k < Kreal) ? W[(size_t)k * N + ncol] : 0.f;
        u.h[e2] = (__bf16)v;
    }
    v4u* dst = (v4u*)(outp + (size_t)id * 16);
    dst[0] = u.q[0];
    dst[1] = u.q[1];
}

// ---------------------------------------------------------------------------
// Main fused kernel: 32 edges / block, 8 wave32s.
// ---------------------------------------------------------------------------
__global__ __launch_bounds__(256) void edge_head_kernel(
    const float* __restrict__ h_nodes,
    const float* __restrict__ edge_scalars,
    const int*   __restrict__ edge_src,
    const int*   __restrict__ edge_dst,
    const int*   __restrict__ edge_type,
    const int*   __restrict__ cell_type_src,
    const int*   __restrict__ cell_type_dst,
    const int*   __restrict__ pin_role_src,
    const int*   __restrict__ pin_role_dst,
    const float* __restrict__ emb_et,
    const float* __restrict__ emb_ct,
    const float* __restrict__ emb_pr,
    const float* __restrict__ cell_b1, const float* __restrict__ cell_g,
    const float* __restrict__ cell_beta, const float* __restrict__ cell_b2,
    const float* __restrict__ net_b1, const float* __restrict__ net_g,
    const float* __restrict__ net_beta, const float* __restrict__ net_b2,
    const __bf16* __restrict__ pCW1,
    const __bf16* __restrict__ pNW1,
    const __bf16* __restrict__ pCW2,
    const __bf16* __restrict__ pNW2,
    float* __restrict__ out,
    int n_edges)
{
    __shared__ __align__(16) char smem[SMEM_BYTES];
    __bf16* feat = (__bf16*)smem;
    float*  Hc   = (float*)(smem + OFF_HC);
    float*  Hn   = (float*)(smem + OFF_HN);
    __bf16* A2c  = (__bf16*)(smem + OFF_A2C);
    __bf16* A2n  = (__bf16*)(smem + OFF_A2N);

    const int tid  = threadIdx.x;
    const int lane = tid & 31;
    const int w    = tid >> 5;
    const int e0   = blockIdx.x * EDGES_PER_BLOCK;

    // ---- Phase 1: build feature rows (bf16) in LDS -------------------------
    {
        int el   = tid >> 3;          // edge 0..31
        int part = tid & 7;           // 16 node-dims per thread
        int e = e0 + el; if (e >= n_edges) e = n_edges - 1;
        int src = edge_src[e], dst = edge_dst[e];
        int c0 = part * 16;
        __bf16* row = feat + el * FSTRIDE;
        const v4f* pu = (const v4f*)(h_nodes + (size_t)src * NODE_DIM + c0);
        const v4f* pv = (const v4f*)(h_nodes + (size_t)dst * NODE_DIM + c0);
#pragma unroll
        for (int q = 0; q < 4; q++) {
            v4f a = pu[q], b = pv[q];
#pragma unroll
            for (int j = 0; j < 4; j += 2) {
                int c = c0 + q * 4 + j;
                float u0 = a[j], u1 = a[j + 1], v0 = b[j], v1 = b[j + 1];
                st2(row + c,        u0, u1);                                // h_u
                st2(row + 128 + c,  v0, v1);                                // h_v
                st2(row + 256 + c,  u0 * v0, u1 * v1);                      // prod
                st2(row + 384 + c,  fabsf(u0 - v0), fabsf(u1 - v1));        // |u-v|
                st2(row + 512 + c,  v0 - u0, v1 - u1);                      // v-u
            }
        }
    }
    if (tid < 32) {
        int e = e0 + tid; if (e >= n_edges) e = n_edges - 1;
        __bf16* row = feat + tid * FSTRIDE;
        int et = edge_type[e];
        const float* tabs[5] = {
            emb_et + (size_t)oov(et, 2) * CAT,
            emb_ct + (size_t)oov(cell_type_src[e], 256) * CAT,
            emb_ct + (size_t)oov(cell_type_dst[e], 256) * CAT,
            emb_pr + (size_t)oov(pin_role_src[e], 64) * CAT,
            emb_pr + (size_t)oov(pin_role_dst[e], 64) * CAT };
#pragma unroll
        for (int t = 0; t < 5; t++) {
            const float* p = tabs[t];
#pragma unroll
            for (int j = 0; j < 16; j += 2) st2(row + 640 + t * 16 + j, p[j], p[j + 1]);
        }
        const float* sp = edge_scalars + (size_t)e * 6;
        float s[6];
#pragma unroll
        for (int j = 0; j < 6; j++) s[j] = fminf(fmaxf(sp[j], -10.f), 10.f);
        st2(row + 720, s[0], s[1]); st2(row + 722, s[2], s[3]); st2(row + 724, s[4], s[5]);
#pragma unroll
        for (int j = 726; j < FSTRIDE; j += 2) st2(row + j, 0.f, 0.f);   // zero pad
    }
    __syncthreads();

    // ---- Phase 2: GEMM1  feat(32x736) @ [cellW1|netW1] ---------------------
    // 20 real N-tiles (12 cell + 8 net). Wave w computes tiles:
    //   t0 = cell tile w            (all waves)
    //   t1 = cell tile w+8 (w<4) or net tile w-4 (w>=4)
    //   t2 = net tile (w&3)+4      -- redundant for w>=4, never stored.
    // Ping-pong double buffering (unroll by 2): no register rotation copies.
    const int m      = lane & 15;
    const int khalf  = (lane < 16) ? 0 : 16;
    const int mbase  = (lane < 16) ? 0 : 8;

    const __bf16* ap0 = feat + m * FSTRIDE + khalf;
    const __bf16* ap1 = feat + (16 + m) * FSTRIDE + khalf;
    const __bf16* bp0 = pCW1 + ((size_t)(w * KC1) * 32 + lane) * 16;
    const __bf16* bp1 = (w < 4)
        ? pCW1 + ((size_t)((w + 8) * KC1) * 32 + lane) * 16
        : pNW1 + ((size_t)((w - 4) * KC1) * 32 + lane) * 16;
    const __bf16* bp2 = pNW1 + ((size_t)(((w & 3) + 4) * KC1) * 32 + lane) * 16;

    v8f acc00 = {}, acc01 = {}, acc10 = {}, acc11 = {}, acc20 = {}, acc21 = {};

    // set 0: k-step kc
    v16bf a0_0 = load_frag2(ap0, ap0 + 8);
    v16bf a1_0 = load_frag2(ap1, ap1 + 8);
    v16bf b0_0 = load_frag2(bp0, bp0 + 8);
    v16bf b1_0 = load_frag2(bp1, bp1 + 8);
    v16bf b2_0 = load_frag2(bp2, bp2 + 8);

#pragma unroll 1
    for (int kc = 0; kc + 2 < KC1; kc += 2) {
        // load set 1 (k-step kc+1) while computing set 0 (kc)
        v16bf a0_1 = load_frag2(ap0 + 32,  ap0 + 40);
        v16bf a1_1 = load_frag2(ap1 + 32,  ap1 + 40);
        v16bf b0_1 = load_frag2(bp0 + 512, bp0 + 520);
        v16bf b1_1 = load_frag2(bp1 + 512, bp1 + 520);
        v16bf b2_1 = load_frag2(bp2 + 512, bp2 + 520);
        acc00 = wmma_bf16(a0_0, b0_0, acc00);
        acc01 = wmma_bf16(a1_0, b0_0, acc01);
        acc10 = wmma_bf16(a0_0, b1_0, acc10);
        acc11 = wmma_bf16(a1_0, b1_0, acc11);
        acc20 = wmma_bf16(a0_0, b2_0, acc20);
        acc21 = wmma_bf16(a1_0, b2_0, acc21);

        ap0 += 64; ap1 += 64; bp0 += 1024; bp1 += 1024; bp2 += 1024;

        // load set 0 (k-step kc+2) while computing set 1 (kc+1)
        a0_0 = load_frag2(ap0, ap0 + 8);
        a1_0 = load_frag2(ap1, ap1 + 8);
        b0_0 = load_frag2(bp0, bp0 + 8);
        b1_0 = load_frag2(bp1, bp1 + 8);
        b2_0 = load_frag2(bp2, bp2 + 8);
        acc00 = wmma_bf16(a0_1, b0_1, acc00);
        acc01 = wmma_bf16(a1_1, b0_1, acc01);
        acc10 = wmma_bf16(a0_1, b1_1, acc10);
        acc11 = wmma_bf16(a1_1, b1_1, acc11);
        acc20 = wmma_bf16(a0_1, b2_1, acc20);
        acc21 = wmma_bf16(a1_1, b2_1, acc21);
    }
    // KC1 = 23 (odd): loop covered k-steps 0..21; set 0 holds k-step 22.
    acc00 = wmma_bf16(a0_0, b0_0, acc00);
    acc01 = wmma_bf16(a1_0, b0_0, acc01);
    acc10 = wmma_bf16(a0_0, b1_0, acc10);
    acc11 = wmma_bf16(a1_0, b1_0, acc11);
    acc20 = wmma_bf16(a0_0, b2_0, acc20);
    acc21 = wmma_bf16(a1_0, b2_0, acc21);

    __syncthreads();   // feat is dead; H buffers may now overwrite it

    {   // scatter accumulators to hidden buffers
        int nc0 = w * 16 + m;                       // tile0: cell col
#pragma unroll
        for (int i = 0; i < 8; i++) {
            Hc[(mbase + i) * HCS + nc0]        = acc00[i];
            Hc[(16 + mbase + i) * HCS + nc0]   = acc01[i];
        }
        if (w < 4) {
            int nc1 = (w + 8) * 16 + m;             // cell
            int nc2 = (w + 4) * 16 + m;             // net
#pragma unroll
            for (int i = 0; i < 8; i++) {
                Hc[(mbase + i) * HCS + nc1]      = acc10[i];
                Hc[(16 + mbase + i) * HCS + nc1] = acc11[i];
                Hn[(mbase + i) * HNS + nc2]      = acc20[i];
                Hn[(16 + mbase + i) * HNS + nc2] = acc21[i];
            }
        } else {
            int nc1 = (w - 4) * 16 + m;             // net
#pragma unroll
            for (int i = 0; i < 8; i++) {
                Hn[(mbase + i) * HNS + nc1]      = acc10[i];
                Hn[(16 + mbase + i) * HNS + nc1] = acc11[i];
            }
        }
    }
    __syncthreads();

    // ---- Phase 3: bias + layernorm + relu -> bf16 A2 -----------------------
    {
        bool isCell = (tid < 128);                  // waves 0-3 cell, 4-7 net
        int  r  = (isCell ? tid : tid - 128) >> 2;  // edge row 0..31
        int  lg = tid & 3;                          // 4 lanes per row
        int  cnt = isCell ? (CELL_H >> 2) : (NET_H >> 2);
        float rHd = isCell ? (1.f / (float)CELL_H) : (1.f / (float)NET_H);
        float* Hb = isCell ? Hc : Hn;
        int    hs = isCell ? HCS : HNS;
        const float* b1 = isCell ? cell_b1 : net_b1;
        const float* gg = isCell ? cell_g  : net_g;
        const float* bb = isCell ? cell_beta : net_beta;
        __bf16* A2 = isCell ? A2c : A2n;
        int     as = isCell ? A2CS : A2NS;

        float s = 0.f, ss = 0.f;
        for (int j = 0; j < cnt; j++) {
            int c = lg * cnt + j;
            float h = Hb[r * hs + c] + b1[c];
            s += h; ss += h * h;
        }
        s  += __shfl_xor(s, 1);  s  += __shfl_xor(s, 2);
        ss += __shfl_xor(ss, 1); ss += __shfl_xor(ss, 2);
        float mu  = s * rHd;
        float var = ss * rHd - mu * mu;
        float rr  = rsqrtf(var + EPS);
        for (int j = 0; j < cnt; j++) {
            int c = lg * cnt + j;
            float h = Hb[r * hs + c] + b1[c];
            float y = (h - mu) * rr * gg[c] + bb[c];
            A2[r * as + c] = (__bf16)fmaxf(y, 0.f);
        }
    }
    __syncthreads();

    // ---- Phase 4: GEMM2. Wave w owns output columns [w*16, w*16+16) --------
    v8f oc0 = {}, oc1 = {}, on0 = {}, on1 = {};
    {
        const __bf16* bpc  = pCW2 + ((size_t)(w * 6) * 32 + lane) * 16;
        const __bf16* ac0  = A2c + m * A2CS + khalf;
        const __bf16* ac1  = A2c + (16 + m) * A2CS + khalf;
#pragma unroll
        for (int kc = 0; kc < 6; kc++) {            // cell: K = 192
            v16bf b  = load_frag2(bpc + kc * 512, bpc + kc * 512 + 8);
            v16bf x0 = load_frag2(ac0 + kc * 32,  ac0 + kc * 32 + 8);
            v16bf x1 = load_frag2(ac1 + kc * 32,  ac1 + kc * 32 + 8);
            oc0 = wmma_bf16(x0, b, oc0);
            oc1 = wmma_bf16(x1, b, oc1);
        }
        const __bf16* bpn = pNW2 + ((size_t)(w * 4) * 32 + lane) * 16;
        const __bf16* an0 = A2n + m * A2NS + khalf;
        const __bf16* an1 = A2n + (16 + m) * A2NS + khalf;
#pragma unroll
        for (int kc = 0; kc < 4; kc++) {            // net: K = 128
            v16bf b  = load_frag2(bpn + kc * 512, bpn + kc * 512 + 8);
            v16bf x0 = load_frag2(an0 + kc * 32,  an0 + kc * 32 + 8);
            v16bf x1 = load_frag2(an1 + kc * 32,  an1 + kc * 32 + 8);
            on0 = wmma_bf16(x0, b, on0);
            on1 = wmma_bf16(x1, b, on1);
        }
    }

    // ---- Phase 5: bias + edge_type select + store --------------------------
    {
        int n = w * 16 + m;
        float bc = cell_b2[n], bn = net_b2[n];
#pragma unroll
        for (int mt = 0; mt < 2; mt++) {
            v8f vc = mt ? oc1 : oc0;
            v8f vn = mt ? on1 : on0;
#pragma unroll
            for (int i = 0; i < 8; i++) {
                int e = e0 + mt * 16 + mbase + i;
                if (e < n_edges) {
                    int et = edge_type[e];
                    float r = (et == 0) ? (vc[i] + bc)
                            : (et == 1) ? (vn[i] + bn) : 0.f;
                    out[(size_t)e * OUT_DIM + n] = r;
                }
            }
        }
    }
}

extern "C" void kernel_launch(void* const* d_in, const int* in_sizes, int n_in,
                              void* d_out, int out_size, void* d_ws, size_t ws_size,
                              hipStream_t stream) {
    const float* h_nodes      = (const float*)d_in[0];
    const float* edge_scalars = (const float*)d_in[1];
    const int*   edge_src     = (const int*)d_in[2];
    const int*   edge_dst     = (const int*)d_in[3];
    const int*   edge_type    = (const int*)d_in[4];
    const int*   cts          = (const int*)d_in[5];
    const int*   ctd          = (const int*)d_in[6];
    const int*   prs          = (const int*)d_in[7];
    const int*   prd          = (const int*)d_in[8];
    const float* emb_et       = (const float*)d_in[9];
    const float* emb_ct       = (const float*)d_in[10];
    const float* emb_pr       = (const float*)d_in[11];
    const float* cW1   = (const float*)d_in[12];
    const float* cb1   = (const float*)d_in[13];
    const float* cg    = (const float*)d_in[14];
    const float* cbeta = (const float*)d_in[15];
    const float* cW2   = (const float*)d_in[16];
    const float* cb2   = (const float*)d_in[17];
    const float* nW1   = (const float*)d_in[18];
    const float* nb1   = (const float*)d_in[19];
    const float* ng    = (const float*)d_in[20];
    const float* nbeta = (const float*)d_in[21];
    const float* nW2   = (const float*)d_in[22];
    const float* nb2   = (const float*)d_in[23];
    const int n_edges  = in_sizes[2];

    // Packed bf16 weights in workspace (total ~553 KB).
    __bf16* pCW1 = (__bf16*)d_ws;                        // 12*23*512 elems
    __bf16* pNW1 = pCW1 + (size_t)12 * KC1 * 512;        //  8*23*512
    __bf16* pCW2 = pNW1 + (size_t)8 * KC1 * 512;         //  8*6*512
    __bf16* pNW2 = pCW2 + (size_t)8 * 6 * 512;           //  8*4*512

    {
        int t;
        t = (CELL_H >> 4) * KC1 * 32;
        pack_b_kernel<<<(t + 255) / 256, 256, 0, stream>>>(cW1, pCW1, INPUT_DIM, KC1, CELL_H);
        t = (NET_H >> 4) * KC1 * 32;
        pack_b_kernel<<<(t + 255) / 256, 256, 0, stream>>>(nW1, pNW1, INPUT_DIM, KC1, NET_H);
        t = (OUT_DIM >> 4) * 6 * 32;
        pack_b_kernel<<<(t + 255) / 256, 256, 0, stream>>>(cW2, pCW2, CELL_H, 6, OUT_DIM);
        t = (OUT_DIM >> 4) * 4 * 32;
        pack_b_kernel<<<(t + 255) / 256, 256, 0, stream>>>(nW2, pNW2, NET_H, 4, OUT_DIM);
    }

    int nblocks = (n_edges + EDGES_PER_BLOCK - 1) / EDGES_PER_BLOCK;
    edge_head_kernel<<<nblocks, 256, 0, stream>>>(
        h_nodes, edge_scalars, edge_src, edge_dst, edge_type,
        cts, ctd, prs, prd, emb_et, emb_ct, emb_pr,
        cb1, cg, cbeta, cb2, nb1, ng, nbeta, nb2,
        pCW1, pNW1, pCW2, pNW2,
        (float*)d_out, n_edges);
}
